// DGMGGraphProp_37555194036643
// MI455X (gfx1250) — matrix operand order repaired and checked by reading, not compiled, not run on hardware
//
#include <hip/hip_runtime.h>
#include <math.h>

// Problem constants (DGMG graph prop): N=50000, E=800000
#define H     128
#define EHD   26
#define INRAW 282   // 2H + EH
#define INPAD 288   // padded to 9 * 32 for K-tiling
#define MH    256
#define OUTD  256   // 2H
#define G3H   384   // 3H
#define TILE_E 128
#define TILE_N 64

typedef __attribute__((ext_vector_type(4)))  __bf16 v4bf;
typedef __attribute__((ext_vector_type(8)))  __bf16 v8bf;
typedef __attribute__((ext_vector_type(16))) __bf16 v16bf;
typedef __attribute__((ext_vector_type(8)))  float  v8f;

__device__ __forceinline__ v16bf cat16(v8bf lo, v8bf hi) {
  return __builtin_shufflevector(lo, hi, 0,1,2,3,4,5,6,7,8,9,10,11,12,13,14,15);
}

__device__ __forceinline__ v4bf cvt4(float4 f) {
  v4bf r;
  r[0] = (__bf16)f.x; r[1] = (__bf16)f.y; r[2] = (__bf16)f.z; r[3] = (__bf16)f.w;
  return r;
}

// A-fragment (16x32 bf16) load from row-major bf16 LDS tile, stride in halfs.
__device__ __forceinline__ v16bf load_afrag(const __bf16* base) {
  v8bf lo = *(const v8bf*)(const void*)base;
  v8bf hi = *(const v8bf*)(const void*)(base + 16);
  return cat16(lo, hi);
}

// ---------------------------------------------------------------------------
// Weight packing: f32 -> bf16 tiles in per-lane B-fragment layout of
// V_WMMA_*_16X16X32: tile (kt,ct) = 512 bf16; lane L holds column ct*16+(L&15),
// K = kt*32 + (L>>4)*8 + e for e<8, kt*32 + 16 + (L>>4)*8 + (e-8) for e>=8.
// transpose=1 reads src as [t][Ncols][Ksrc] (for GRU W^T matrices).
// ---------------------------------------------------------------------------
__global__ void pack_weights_kernel(const float* __restrict__ src, __bf16* __restrict__ dst,
                                    int Ksrc, int Ncols, int KT, int CT, int transpose, int R) {
  int idx = blockIdx.x * blockDim.x + threadIdx.x;
  int total = R * KT * CT * 512;
  if (idx >= total) return;
  int e    = idx & 15;
  int lane = (idx >> 4) & 31;
  int rest = idx >> 9;
  int ct = rest % CT; rest /= CT;
  int kt = rest % KT;
  int t  = rest / KT;
  int k   = kt * 32 + (lane >> 4) * 8 + (e & 7) + (e >> 3) * 16;
  int col = ct * 16 + (lane & 15);
  float v = 0.0f;
  if (k < Ksrc && col < Ncols) {
    if (transpose) v = src[(size_t)t * Ncols * Ksrc + (size_t)col * Ksrc + k];
    else           v = src[(size_t)t * Ksrc * Ncols + (size_t)k * Ncols + col];
  }
  dst[idx] = (__bf16)v;
}

// ---------------------------------------------------------------------------
// Edge MLP: per 128-edge tile, gather concat(hv[dst],hv[src],he) -> bf16 LDS,
// h1 = relu(x@W1+b1) via bf16 WMMA, m = h1@W2+b2, atomic scatter-add to a[dst].
// 8 waves; wave w owns row-tile w and keeps its A fragments in registers,
// looping only over the 16 column tiles (B comes from L2-resident packed
// weights). Wave-uniform control flow keeps EXEC all-ones for WMMA.
// ---------------------------------------------------------------------------
__global__ void edge_mlp_kernel(const float* __restrict__ hv, const float* __restrict__ he,
                                const int* __restrict__ src, const int* __restrict__ dst,
                                const __bf16* __restrict__ w1p, const __bf16* __restrict__ w2p,
                                const float* __restrict__ b1, const float* __restrict__ b2,
                                float* __restrict__ abuf, int t, int E) {
  extern __shared__ __align__(16) char smem[];
  __bf16* xs   = (__bf16*)smem;                      // [128][288] bf16
  __bf16* h1s  = xs + TILE_E * INPAD;                // [128][256] bf16
  int*    dsts = (int*)(void*)(h1s + TILE_E * MH);   // [128]
  int*    srcs = dsts + TILE_E;                      // [128]

  const int tid = threadIdx.x;
  const int e0  = blockIdx.x * TILE_E;

  { // stage edge indices
    int r = tid & (TILE_E - 1);
    int e = e0 + r; if (e >= E) e = E - 1;
    if (tid < TILE_E) dsts[r] = dst[e];
    else              srcs[r] = src[e];
  }
  __syncthreads();

  { // vectorized gather: [128][288] in float4 chunks -> packed bf16 ds_store_b64
    for (int v = tid; v < TILE_E * (INPAD / 4); v += 256) {
      int row = v / (INPAD / 4);
      int c   = (v % (INPAD / 4)) * 4;
      v4bf o;
      if (c < H) {
        o = cvt4(*(const float4*)(const void*)(hv + (size_t)dsts[row] * H + c));
      } else if (c < 2 * H) {
        o = cvt4(*(const float4*)(const void*)(hv + (size_t)srcs[row] * H + (c - H)));
      } else {
        int ee = e0 + row; if (ee >= E) ee = E - 1;
        const float* hee = he + (size_t)ee * EHD;
        #pragma unroll
        for (int i = 0; i < 4; ++i) {
          int k = c - 2 * H + i;
          o[i] = (__bf16)((k < EHD) ? hee[k] : 0.0f);
        }
      }
      *(v4bf*)(void*)(xs + row * INPAD + c) = o;
    }
  }
  __syncthreads();

  const int lane = tid & 31, wid = tid >> 5;
  const int nq = lane & 15, hq = lane >> 4;
  const int rt = wid;                    // wave owns one 16-edge row-tile

  // ---- layer 1: [128 x 288] @ [288 x 256] ----
  {
    v16bf afr[9];
    #pragma unroll
    for (int kt = 0; kt < 9; ++kt)
      afr[kt] = load_afrag(xs + (rt * 16 + nq) * INPAD + kt * 32 + hq * 8);

    for (int ct = 0; ct < 16; ++ct) {
      v8f acc = {0.f,0.f,0.f,0.f,0.f,0.f,0.f,0.f};
      #pragma unroll
      for (int kt = 0; kt < 9; ++kt) {
        v16bf B = *(const v16bf*)(const void*)(w1p + ((((size_t)t * 9 + kt) * 16 + ct) << 9) + lane * 16);
        acc = __builtin_amdgcn_wmma_f32_16x16x32_bf16(false, afr[kt], false, B, (short)0, acc, false, false);
      }
      float bb = b1[t * MH + ct * 16 + nq];
      #pragma unroll
      for (int v = 0; v < 8; ++v) {
        float r = acc[v] + bb;
        r = r > 0.0f ? r : 0.0f;
        h1s[(rt * 16 + hq * 8 + v) * MH + ct * 16 + nq] = (__bf16)r;
      }
    }
  }
  __syncthreads();

  // ---- layer 2: [128 x 256] @ [256 x 256] + atomic scatter to a[dst] ----
  {
    v16bf afr[8];
    #pragma unroll
    for (int kt = 0; kt < 8; ++kt)
      afr[kt] = load_afrag(h1s + (rt * 16 + nq) * MH + kt * 32 + hq * 8);

    for (int ct = 0; ct < 16; ++ct) {
      v8f acc = {0.f,0.f,0.f,0.f,0.f,0.f,0.f,0.f};
      #pragma unroll
      for (int kt = 0; kt < 8; ++kt) {
        v16bf B = *(const v16bf*)(const void*)(w2p + ((((size_t)t * 8 + kt) * 16 + ct) << 9) + lane * 16);
        acc = __builtin_amdgcn_wmma_f32_16x16x32_bf16(false, afr[kt], false, B, (short)0, acc, false, false);
      }
      float bb = b2[t * OUTD + ct * 16 + nq];
      #pragma unroll
      for (int v = 0; v < 8; ++v) {
        int er = rt * 16 + hq * 8 + v;
        if (e0 + er < E) {
          int node = dsts[er];
          unsafeAtomicAdd(abuf + (size_t)node * OUTD + ct * 16 + nq, acc[v] + bb);
        }
      }
    }
  }
}

// ---------------------------------------------------------------------------
// GRU: per 64-node tile, gx = a@wih^T, gh = hv@whh^T via bf16 WMMA.
// Wave w owns (rt = w&3, half = w>>2); holds 8 a-frags + 4 hv-frags in
// registers and covers 12 j-tiles (both halves = 144 WMMA, balanced).
// r/z parts summed; n-gate gx/gh kept separate (n = tanh(gx_n + r*gh_n)).
// ---------------------------------------------------------------------------
__global__ void gru_kernel(const float* __restrict__ abuf, const float* __restrict__ hvin,
                           const __bf16* __restrict__ wihp, const __bf16* __restrict__ whhp,
                           const float* __restrict__ bih, const float* __restrict__ bhh,
                           float* __restrict__ hvout, int t, int N) {
  extern __shared__ __align__(16) char smem[];
  __bf16* as = (__bf16*)smem;                        // [64][256] bf16
  __bf16* hs = as + TILE_N * OUTD;                   // [64][128] bf16
  float* grz = (float*)(void*)(hs + TILE_N * H);     // [64][256] f32 (r|z)
  float* gxn = grz + TILE_N * OUTD;                  // [64][128] f32
  float* ghn = gxn + TILE_N * H;                     // [64][128] f32

  const int tid = threadIdx.x;
  const int n0  = blockIdx.x * TILE_N;

  { // vectorized gather of a (f32->bf16) and hv (f32->bf16)
    for (int v = tid; v < TILE_N * ((OUTD + H) / 4); v += 256) {
      int row = v / ((OUTD + H) / 4);
      int c   = (v % ((OUTD + H) / 4)) * 4;
      int node = n0 + row; if (node >= N) node = N - 1;
      if (c < OUTD) {
        v4bf o = cvt4(*(const float4*)(const void*)(abuf + (size_t)node * OUTD + c));
        *(v4bf*)(void*)(as + row * OUTD + c) = o;
      } else {
        v4bf o = cvt4(*(const float4*)(const void*)(hvin + (size_t)node * H + (c - OUTD)));
        *(v4bf*)(void*)(hs + row * H + (c - OUTD)) = o;
      }
    }
  }
  __syncthreads();

  const int lane = tid & 31, wid = tid >> 5;
  const int nq = lane & 15, hq = lane >> 4;
  const int rt   = wid & 3;
  const int half = wid >> 2;

  {
    v16bf af[8];
    #pragma unroll
    for (int kt = 0; kt < 8; ++kt)
      af[kt] = load_afrag(as + (rt * 16 + nq) * OUTD + kt * 32 + hq * 8);
    v16bf hf[4];
    #pragma unroll
    for (int kt = 0; kt < 4; ++kt)
      hf[kt] = load_afrag(hs + (rt * 16 + nq) * H + kt * 32 + hq * 8);

    for (int jj = 0; jj < 12; ++jj) {
      int jt = half * 12 + jj;           // wave-uniform
      int j  = jt * 16 + nq;
      if (jt < 16) {                     // r|z block: gx + gh summed
        v8f acc = {0.f,0.f,0.f,0.f,0.f,0.f,0.f,0.f};
        #pragma unroll
        for (int kt = 0; kt < 8; ++kt) {
          v16bf B = *(const v16bf*)(const void*)(wihp + ((((size_t)t * 8 + kt) * 24 + jt) << 9) + lane * 16);
          acc = __builtin_amdgcn_wmma_f32_16x16x32_bf16(false, af[kt], false, B, (short)0, acc, false, false);
        }
        #pragma unroll
        for (int kt = 0; kt < 4; ++kt) {
          v16bf B = *(const v16bf*)(const void*)(whhp + ((((size_t)t * 4 + kt) * 24 + jt) << 9) + lane * 16);
          acc = __builtin_amdgcn_wmma_f32_16x16x32_bf16(false, hf[kt], false, B, (short)0, acc, false, false);
        }
        float bias = bih[t * G3H + j] + bhh[t * G3H + j];
        #pragma unroll
        for (int v = 0; v < 8; ++v)
          grz[(rt * 16 + hq * 8 + v) * OUTD + jt * 16 + nq] = acc[v] + bias;
      } else {                           // n block: keep gx_n, gh_n separate
        v8f accx = {0.f,0.f,0.f,0.f,0.f,0.f,0.f,0.f};
        v8f acch = {0.f,0.f,0.f,0.f,0.f,0.f,0.f,0.f};
        #pragma unroll
        for (int kt = 0; kt < 8; ++kt) {
          v16bf B = *(const v16bf*)(const void*)(wihp + ((((size_t)t * 8 + kt) * 24 + jt) << 9) + lane * 16);
          accx = __builtin_amdgcn_wmma_f32_16x16x32_bf16(false, af[kt], false, B, (short)0, accx, false, false);
        }
        #pragma unroll
        for (int kt = 0; kt < 4; ++kt) {
          v16bf B = *(const v16bf*)(const void*)(whhp + ((((size_t)t * 4 + kt) * 24 + jt) << 9) + lane * 16);
          acch = __builtin_amdgcn_wmma_f32_16x16x32_bf16(false, hf[kt], false, B, (short)0, acch, false, false);
        }
        float bx = bih[t * G3H + j];
        float bh = bhh[t * G3H + j];
        #pragma unroll
        for (int v = 0; v < 8; ++v) {
          int row = rt * 16 + hq * 8 + v;
          gxn[row * H + (jt - 16) * 16 + nq] = accx[v] + bx;
          ghn[row * H + (jt - 16) * 16 + nq] = acch[v] + bh;
        }
      }
    }
  }
  __syncthreads();

  // elementwise GRU epilogue
  for (int i = tid; i < TILE_N * H; i += 256) {
    int row = i >> 7, c = i & (H - 1);
    int node = n0 + row;
    if (node < N) {
      float r = 1.0f / (1.0f + expf(-grz[row * OUTD + c]));
      float z = 1.0f / (1.0f + expf(-grz[row * OUTD + H + c]));
      float n = tanhf(gxn[row * H + c] + r * ghn[row * H + c]);
      float ho = hvin[(size_t)node * H + c];
      hvout[(size_t)node * H + c] = (1.0f - z) * n + z * ho;
    }
  }
}

// ---------------------------------------------------------------------------
extern "C" void kernel_launch(void* const* d_in, const int* in_sizes, int n_in,
                              void* d_out, int out_size, void* d_ws, size_t ws_size,
                              hipStream_t stream) {
  const float* hv  = (const float*)d_in[0];
  const float* he  = (const float*)d_in[1];
  const int*   src = (const int*)d_in[2];
  const int*   dst = (const int*)d_in[3];
  const float* w1  = (const float*)d_in[4];
  const float* b1  = (const float*)d_in[5];
  const float* w2  = (const float*)d_in[6];
  const float* b2  = (const float*)d_in[7];
  const float* wih = (const float*)d_in[8];
  const float* whh = (const float*)d_in[9];
  const float* bih = (const float*)d_in[10];
  const float* bhh = (const float*)d_in[11];
  float* out = (float*)d_out;

  const int N = in_sizes[0] / H;            // 50000
  const int E = in_sizes[2];                // 800000
  const int R = in_sizes[4] / (INRAW * MH); // 2

  char* ws = (char*)d_ws;
  size_t off = 0;
  auto suballoc = [&](size_t bytes) -> void* {
    void* p = ws + off;
    off = (off + bytes + 255) & ~(size_t)255;
    return p;
  };
  float*  abuf  = (float*)suballoc((size_t)N * OUTD * sizeof(float));
  float*  hvbuf = (float*)suballoc((size_t)N * H * sizeof(float));
  __bf16* w1p   = (__bf16*)suballoc((size_t)R * 9 * 16 * 512 * sizeof(__bf16));
  __bf16* w2p   = (__bf16*)suballoc((size_t)R * 8 * 16 * 512 * sizeof(__bf16));
  __bf16* wihp  = (__bf16*)suballoc((size_t)R * 8 * 24 * 512 * sizeof(__bf16));
  __bf16* whhp  = (__bf16*)suballoc((size_t)R * 4 * 24 * 512 * sizeof(__bf16));

  // pack weights (bf16, B-fragment tile layout)
  {
    int tot;
    tot = R * 9 * 16 * 512;
    pack_weights_kernel<<<(tot + 255) / 256, 256, 0, stream>>>(w1, w1p, INRAW, MH, 9, 16, 0, R);
    tot = R * 8 * 16 * 512;
    pack_weights_kernel<<<(tot + 255) / 256, 256, 0, stream>>>(w2, w2p, MH, OUTD, 8, 16, 0, R);
    tot = R * 8 * 24 * 512;
    pack_weights_kernel<<<(tot + 255) / 256, 256, 0, stream>>>(wih, wihp, OUTD, G3H, 8, 24, 1, R);
    tot = R * 4 * 24 * 512;
    pack_weights_kernel<<<(tot + 255) / 256, 256, 0, stream>>>(whh, whhp, H, G3H, 4, 24, 1, R);
  }

  const size_t edge_lds = (size_t)TILE_E * INPAD * 2 + (size_t)TILE_E * MH * 2 + 2 * TILE_E * 4;
  const size_t gru_lds  = (size_t)TILE_N * OUTD * 2 + (size_t)TILE_N * H * 2 +
                          (size_t)TILE_N * OUTD * 4 + (size_t)TILE_N * H * 4 * 2;
  const int eblocks = (E + TILE_E - 1) / TILE_E;
  const int nblocks = (N + TILE_N - 1) / TILE_N;

  const float* hv_cur = hv;
  for (int t = 0; t < R; ++t) {
    hipMemsetAsync(abuf, 0, (size_t)N * OUTD * sizeof(float), stream);
    edge_mlp_kernel<<<eblocks, 256, edge_lds, stream>>>(hv_cur, he, src, dst,
                                                        w1p, w2p, b1, b2, abuf, t, E);
    float* hv_next = (t == R - 1) ? out : hvbuf;
    gru_kernel<<<nblocks, 256, gru_lds, stream>>>(abuf, hv_cur, wihp, whhp,
                                                  bih, bhh, hv_next, t, N);
    hv_cur = hv_next;
  }
}
